// SlidingWindowAttention_42073499631647
// MI455X (gfx1250) — compile-verified, matrix operand order checked by reading, not run on hardware
//
#include <hip/hip_runtime.h>

// ---------------------------------------------------------------------------
// CDNA5 (gfx1250) sliding-window attention with persistent memory tokens.
// wave32, WMMA bf16 16x16x32 (f32 accum), async-LDS staging + TDM tile loads.
// ---------------------------------------------------------------------------

typedef __attribute__((ext_vector_type(16))) __bf16 v16bf;
typedef __attribute__((ext_vector_type(8)))  float  v8f;
typedef int      v4i_gcc __attribute__((vector_size(16)));  // matches builtin protos
typedef unsigned v4u_gcc __attribute__((vector_size(16)));
typedef int      v8i_gcc __attribute__((vector_size(32)));

#define WMMA_BF16(a, b, c) \
  __builtin_amdgcn_wmma_f32_16x16x32_bf16(false, (a), false, (b), (short)0, (c), false, false)

// Constants of the reference problem
#define BATCH      2
#define SEQ        2048
#define DIM        2048
#define HEADS      16
#define DIM_HEAD   128
#define WINDOW     512
#define NPERSIST   16
#define ROWS       (BATCH * SEQ)       // 4096
#define QKV_COLS   (3 * DIM)           // 6144

// ---- optional CDNA5 async global->LDS path (ASYNCcnt) ---------------------
#if defined(__has_builtin)
#if __has_builtin(__builtin_amdgcn_global_load_async_to_lds_b128) && \
    __has_builtin(__builtin_amdgcn_s_wait_asynccnt)
#define USE_ASYNC_LDS 1
#endif
#if __has_builtin(__builtin_amdgcn_tensor_load_to_lds) && \
    __has_builtin(__builtin_amdgcn_s_wait_tensorcnt)
#define USE_TDM 1
#endif
#endif
#ifndef USE_ASYNC_LDS
#define USE_ASYNC_LDS 0
#endif
#ifndef USE_TDM
#define USE_TDM 0
#endif
#if __clang_major__ >= 23
#define TDM_ARGS6 1          // therock/clang-23: 6-arg form
#else
#define TDM_ARGS6 0          // ROCm 7.2/clang-22: 5-arg form
#endif

#define AS1 __attribute__((address_space(1)))
#define AS3 __attribute__((address_space(3)))

// copy 16 bytes global(bf16) -> LDS, async if available
__device__ __forceinline__ void stage16(void* lds_dst, const void* g_src) {
#if USE_ASYNC_LDS
  __builtin_amdgcn_global_load_async_to_lds_b128(
      (AS1 v4i_gcc*)(uintptr_t)g_src,
      (AS3 v4i_gcc*)(unsigned)(uintptr_t)lds_dst,   // generic LDS low 32 bits = offset
      0, 0);
#else
  *(uint4*)lds_dst = *(const uint4*)g_src;
#endif
}
__device__ __forceinline__ void stage_wait() {
#if USE_ASYNC_LDS
  __builtin_amdgcn_s_wait_asynccnt(0);
#endif
}

#if USE_TDM
// Tensor Data Mover: 2D tile (tile_d1 rows x tile_d0 elems, 2-byte data) from a
// row-major global tensor into contiguous LDS. D# packing per ISA 08 S8.3/8.4.
__device__ __forceinline__ void tdm_load_to_lds(unsigned lds_off, const void* gaddr,
                                                unsigned tensor_d0, unsigned tensor_d1,
                                                unsigned tile_d0, unsigned tile_d1,
                                                unsigned stride_d0) {
  const unsigned long long ga = (unsigned long long)(uintptr_t)gaddr;
  const v4u_gcc g0 = {
      1u,                                              // count=1, user descriptor
      lds_off,                                         // lds_addr (bytes)
      (unsigned)ga,                                    // global_addr[31:0]
      ((unsigned)(ga >> 32) & 0x01ffffffu) | 0x80000000u };  // addr[56:32] | type=2
  const v8i_gcc g1 = {
      (int)(1u << 16),                                 // data_size=1 (2 bytes), mask=0
      (int)((tensor_d0 & 0xffffu) << 16),              // tensor_dim0[15:0] @ bits 63:48
      (int)(((tensor_d0 >> 16) & 0xffffu) | ((tensor_d1 & 0xffffu) << 16)),
      (int)(((tensor_d1 >> 16) & 0xffffu) | ((tile_d0 & 0xffffu) << 16)),
      (int)(tile_d1 & 0xffffu),                        // tile_dim1 (tile_dim2=0)
      (int)stride_d0,                                  // tensor_dim0_stride[31:0]
      0, 0 };
  const v4i_gcc gz = {0, 0, 0, 0};
#if TDM_ARGS6
  const v8i_gcc gz8 = {0, 0, 0, 0, 0, 0, 0, 0};
  __builtin_amdgcn_tensor_load_to_lds(g0, g1, gz, gz, gz8, 0);
#else
  __builtin_amdgcn_tensor_load_to_lds(g0, g1, gz, gz, 0);
#endif
}
#endif

// ---------------------------------------------------------------------------
// WMMA fragment loaders (16x16x32 bf16), wave32.
// A (16x32, MxK): lane m = lane%16, half = lane/16; VGPR v holds bf16 pair
//   k = (v>=4 ? 16 : 0) + half*8 + (v&3)*2      (ISA 7.12.2)
// C/D (16x16): lane holds N = lane%16, VGPR r holds M = r + 8*(lane/16)
// ---------------------------------------------------------------------------
__device__ __forceinline__ v16bf frag_a(const __bf16* p, int ld, int lane) {
  const int m = lane & 15, hf = lane >> 4;
  v16bf f;
#pragma unroll
  for (int v = 0; v < 8; ++v) {
    const int k = ((v & 4) << 2) + hf * 8 + ((v & 3) << 1);
    f[2 * v]     = p[m * ld + k];
    f[2 * v + 1] = p[m * ld + k + 1];
  }
  return f;
}

// B (32x16, KxN) where source tile is stored N-major (rows = n, length ld):
// element (k, n) at p[n*ld + k] -> per-lane 8-element k-runs are contiguous.
__device__ __forceinline__ v16bf frag_bT(const __bf16* p, int ld, int lane) {
  const int n = lane & 15, hf = lane >> 4;
  v16bf f;
#pragma unroll
  for (int v = 0; v < 8; ++v) {
    const int k = ((v & 4) << 2) + hf * 8 + ((v & 3) << 1);
    f[2 * v]     = p[n * ld + k];
    f[2 * v + 1] = p[n * ld + k + 1];
  }
  return f;
}

// B (32x16, KxN) from pair-packed tile: element (k, n) at p[(k/8)*gstride + n*8 + k%8]
// -> each lane-half's 8-element k-run is one contiguous 16B chunk (2 x b128 total).
__device__ __forceinline__ v16bf frag_b_pk(const __bf16* p, int gstride, int lane) {
  const int n = lane & 15, hf = lane >> 4;
  union { v16bf v; uint4 q[2]; } u;
  u.q[0] = *(const uint4*)(p + (size_t)(0 + hf) * gstride + n * 8);
  u.q[1] = *(const uint4*)(p + (size_t)(2 + hf) * gstride + n * 8);
  return u.v;
}

// ---------------------------------------------------------------------------
// Kernel 1: RMSNorm, f32 in -> bf16 normalized activations
// ---------------------------------------------------------------------------
__global__ __launch_bounds__(256)
void rmsnorm_kernel(const float* __restrict__ x, const float* __restrict__ w,
                    __bf16* __restrict__ xn) {
  const int row = blockIdx.x;
  const float* xr = x + (size_t)row * DIM;
  float ss = 0.0f;
  for (int c = threadIdx.x; c < DIM; c += 256) {
    const float v = xr[c];
    ss += v * v;
  }
  __shared__ float red[8];
#pragma unroll
  for (int off = 16; off > 0; off >>= 1) ss += __shfl_down(ss, off, 32);
  if ((threadIdx.x & 31) == 0) red[threadIdx.x >> 5] = ss;
  __syncthreads();
  if (threadIdx.x == 0) {
    float t = 0.0f;
#pragma unroll
    for (int i = 0; i < 8; ++i) t += red[i];
    red[0] = rsqrtf(t * (1.0f / (float)DIM) + 1.1920929e-07f);
  }
  __syncthreads();
  const float sc = red[0];
  for (int c = threadIdx.x; c < DIM; c += 256)
    xn[(size_t)row * DIM + c] = (__bf16)(xr[c] * sc * w[c]);
}

// ---------------------------------------------------------------------------
// Kernel 2/5: C[M,N](f32) = A[M,K](bf16) x B[K,N](f32 -> bf16 on stage)
// Block tile 128x64, K step 32. B tile staged pair-packed: Bs[g][n][8], k=g*8+i.
// N/K are compile-time so B row addresses fold into immediate IOFFSETs.
// ---------------------------------------------------------------------------
template <int N, int K>
__global__ __launch_bounds__(256)
void gemm_bf16f32(const __bf16* __restrict__ A, const float* __restrict__ B,
                  float* __restrict__ C) {
  __shared__ __bf16 As[128][32];
  __shared__ __bf16 Bs[4][64][8];
  const int tid  = threadIdx.x;
  const int lane = tid & 31;
  const int wave = tid >> 5;
  const int wr   = wave & 3;
  const int wc   = wave >> 2;
  const int bm   = blockIdx.y * 128;
  const int bn   = blockIdx.x * 64;

  v8f acc[2][2] = {};

  // staging roles (fixed per thread)
  const int arow = tid >> 1, acol = (tid & 1) << 4;       // A: 2 x 16B
  const int bn_l = tid & 63, bg = tid >> 6;               // B: one packed column

  for (int k0 = 0; k0 < K; k0 += 32) {
    // stage A tile (bf16 128x32)
    {
      const __bf16* s = A + (size_t)(bm + arow) * K + k0 + acol;
      stage16(&As[arow][acol], s);
      stage16(&As[arow][acol + 8], s + 8);
    }
    // stage B tile (f32 -> bf16, pair-packed): 8 coalesced row reads (imm offsets)
    {
      const float* s = B + (size_t)(k0 + bg * 8) * N + bn + bn_l;
      __bf16 tmp[8];
#pragma unroll
      for (int jj = 0; jj < 8; ++jj)
        tmp[jj] = (__bf16)s[(size_t)jj * N];
      *(uint4*)&Bs[bg][bn_l][0] = *(const uint4*)tmp;
    }
    if (k0 + 32 < K) {  // hint next weight tile into cache (global_prefetch_b8)
      __builtin_prefetch(B + (size_t)(k0 + 32 + bg * 8) * N + bn + bn_l, 0, 1);
    }
    stage_wait();
    __syncthreads();

    v16bf af[2], bb[2];
#pragma unroll
    for (int i = 0; i < 2; ++i) af[i] = frag_a(&As[wr * 32 + i * 16][0], 32, lane);
#pragma unroll
    for (int j = 0; j < 2; ++j) bb[j] = frag_b_pk(&Bs[0][wc * 32 + j * 16][0], 64 * 8, lane);
#pragma unroll
    for (int i = 0; i < 2; ++i)
#pragma unroll
      for (int j = 0; j < 2; ++j)
        acc[i][j] = WMMA_BF16(af[i], bb[j], acc[i][j]);
    __syncthreads();
  }

  const int hf = lane >> 4, nc = lane & 15;
#pragma unroll
  for (int i = 0; i < 2; ++i)
#pragma unroll
    for (int j = 0; j < 2; ++j)
#pragma unroll
      for (int r = 0; r < 8; ++r)
        C[(size_t)(bm + wr * 32 + i * 16 + r + 8 * hf) * N + bn + wc * 32 + j * 16 + nc] =
            acc[i][j][r];
}

// ---------------------------------------------------------------------------
// Kernel 3: rotary embedding (interleaved pairs) + head repack.
// q/k -> [b,h,n,128] row-major bf16 (1/sqrt(d) folded into q).
// v    -> pair-packed per 32-key block: [b,h][kb][g][d][i], key = kb*32+g*8+i.
// ---------------------------------------------------------------------------
__global__ __launch_bounds__(64)
void rope_repack_kernel(const float* __restrict__ qkv,
                        __bf16* __restrict__ qr, __bf16* __restrict__ kr,
                        __bf16* __restrict__ vr) {
  const int n = blockIdx.x, h = blockIdx.y, b = blockIdx.z;
  const int p = threadIdx.x;                              // pair index 0..63
  const size_t row = (size_t)(b * SEQ + n) * QKV_COLS;
  const size_t head_off = (size_t)(b * HEADS + h) * SEQ * DIM_HEAD;
  const size_t dst = head_off + (size_t)n * DIM_HEAD;

  const float inv_freq = __expf(-(float)(2 * p) * (1.0f / (float)DIM_HEAD) * 9.210340371976184f);
  float s, c;
  __sincosf((float)n * inv_freq, &s, &c);
  const float scale = 0.08838834764831845f;               // 1/sqrt(128)

  const float q0 = qkv[row + h * DIM_HEAD + 2 * p];
  const float q1 = qkv[row + h * DIM_HEAD + 2 * p + 1];
  qr[dst + 2 * p]     = (__bf16)((q0 * c - q1 * s) * scale);
  qr[dst + 2 * p + 1] = (__bf16)((q1 * c + q0 * s) * scale);

  const float k0 = qkv[row + DIM + h * DIM_HEAD + 2 * p];
  const float k1 = qkv[row + DIM + h * DIM_HEAD + 2 * p + 1];
  kr[dst + 2 * p]     = (__bf16)(k0 * c - k1 * s);
  kr[dst + 2 * p + 1] = (__bf16)(k1 * c + k0 * s);

  // pair-packed V
  const size_t vbase = head_off + (size_t)(n >> 5) * 4096 +
                       (size_t)((n >> 3) & 3) * 1024 + (n & 7);
  vr[vbase + (size_t)p * 8]        = (__bf16)qkv[row + 2 * DIM + h * DIM_HEAD + p];
  vr[vbase + (size_t)(p + 64) * 8] = (__bf16)qkv[row + 2 * DIM + h * DIM_HEAD + p + 64];
}

// ---------------------------------------------------------------------------
// Kernel 4: flash-style sliding-window attention with persistent tokens.
// Block: (query-tile of 128, head, batch); 8 waves, 16 query rows per wave.
// Key tiles of 32 staged in LDS via TDM (fallback: async b128 copies).
// ---------------------------------------------------------------------------
__global__ __launch_bounds__(256)
void attn_kernel(const __bf16* __restrict__ qr, const __bf16* __restrict__ kr,
                 const __bf16* __restrict__ vr, const float* __restrict__ pmem,
                 __bf16* __restrict__ ao) {
  const int qt = blockIdx.x;   // 0..15 : query tile of 128
  const int h  = blockIdx.y;   // head
  const int b  = blockIdx.z;   // batch

  __shared__ __bf16 Ks[32][DIM_HEAD];     // row-major (keys x dims)
  __shared__ __bf16 VsPk[4096];           // pair-packed [g][d][i], key = g*8+i
  __shared__ __bf16 Ps[8][16][32];        // wave-private P (16x32) scratch

  const int tid  = threadIdx.x;
  const int lane = tid & 31;
  const int wave = tid >> 5;
  const int hf   = lane >> 4;
  const int nc   = lane & 15;

  const size_t head_off = (size_t)(b * HEADS + h) * SEQ * DIM_HEAD;
  const int qbase = qt * 128 + wave * 16;

  // Q fragments: 16 rows x 128 dims = 4 A-fragments of 16x32 (scale pre-folded)
  v16bf qf[4];
  {
    const __bf16* qp = qr + head_off + (size_t)qbase * DIM_HEAD;
#pragma unroll
    for (int cchunk = 0; cchunk < 4; ++cchunk)
      qf[cchunk] = frag_a(qp + cchunk * 32, DIM_HEAD, lane);
  }

  float m_run[8], l_run[8];
  v8f O[8] = {};
#pragma unroll
  for (int r = 0; r < 8; ++r) { m_run[r] = -1.0e30f; l_run[r] = 0.0f; }

  int t0 = qt * 128 - WINDOW;
  if (t0 < 0) t0 = 0;
  const int ntiles = (qt * 128 + 128 - t0) / 32;  // real-key tiles of 32

  for (int tile = -1; tile < ntiles; ++tile) {
    // ---- stage K/V tile into LDS ----
    if (tile < 0) {
      // persistent-memory tokens (f32 -> bf16); pad rows 16..31 with zeros
      const int row = tid >> 3;             // 0..31
      const int seg = (tid & 7) << 4;       // 0,16,..,112 (dim offset)
      const float* pk = pmem + ((size_t)h * NPERSIST + (row & 15)) * DIM_HEAD;
      const float* pv = pmem + (size_t)HEADS * NPERSIST * DIM_HEAD +
                        ((size_t)h * NPERSIST + (row & 15)) * DIM_HEAD;
      const bool valid = row < NPERSIST;
      const int vg = (row >> 3) & 3, vi = row & 7;
#pragma unroll
      for (int e = 0; e < 16; ++e) {
        Ks[row][seg + e] = valid ? (__bf16)pk[seg + e] : (__bf16)0.0f;
        VsPk[vg * 1024 + (seg + e) * 8 + vi] = valid ? (__bf16)pv[seg + e] : (__bf16)0.0f;
      }
    } else {
      const int tb = t0 + tile * 32;
#if USE_TDM
      if (wave == 0) {
        // K: 2D tile, 32 rows x 128 bf16, row-major -> contiguous LDS
        tdm_load_to_lds((unsigned)(uintptr_t)&Ks[0][0],
                        kr + head_off + (size_t)tb * DIM_HEAD,
                        DIM_HEAD, SEQ, DIM_HEAD, 32, DIM_HEAD);
        // V: flat 8KB pair-packed block (1D tile)
        tdm_load_to_lds((unsigned)(uintptr_t)&VsPk[0],
                        vr + head_off + (size_t)(tb >> 5) * 4096,
                        4096, 1, 4096, 1, 4096);
        __builtin_amdgcn_s_wait_tensorcnt(0);
      }
#else
      const int row = tid >> 3;
      const int seg = (tid & 7) << 4;
      const __bf16* sk = kr + head_off + (size_t)(tb + row) * DIM_HEAD + seg;
      stage16(&Ks[row][seg], sk);
      stage16(&Ks[row][seg + 8], sk + 8);
      const __bf16* sv = vr + head_off + (size_t)(tb >> 5) * 4096 + tid * 16;
      stage16(&VsPk[tid * 16], sv);
      stage16(&VsPk[tid * 16 + 8], sv + 8);
#endif
    }
    stage_wait();
    __syncthreads();

    // ---- S = Q . K^T  (two 16x16 C tiles covering 32 keys) ----
    v8f s0 = {}, s1 = {};
#pragma unroll
    for (int cchunk = 0; cchunk < 4; ++cchunk) {
      const v16bf bk0 = frag_bT(&Ks[0][cchunk * 32], DIM_HEAD, lane);
      const v16bf bk1 = frag_bT(&Ks[16][cchunk * 32], DIM_HEAD, lane);
      s0 = WMMA_BF16(qf[cchunk], bk0, s0);
      s1 = WMMA_BF16(qf[cchunk], bk1, s1);
    }

    // ---- mask + online softmax (rows live in 16-lane halves) ----
    float p0[8], p1[8];
#pragma unroll
    for (int r = 0; r < 8; ++r) {
      const int i = qbase + r + 8 * hf;     // query position
      float v0 = s0[r], v1 = s1[r];
      if (tile < 0) {
        v1 = -1.0e30f;                       // pad half of persist tile
      } else {
        const int tk0 = t0 + tile * 32 + nc;
        const int tk1 = tk0 + 16;
        if (!(i >= tk0 && (i - tk0) <= WINDOW)) v0 = -1.0e30f;
        if (!(i >= tk1 && (i - tk1) <= WINDOW)) v1 = -1.0e30f;
      }
      float m = fmaxf(v0, v1);
#pragma unroll
      for (int off = 8; off > 0; off >>= 1) m = fmaxf(m, __shfl_xor(m, off, 16));
      m = fmaxf(m, m_run[r]);
      const float alpha = __expf(m_run[r] - m);
      p0[r] = __expf(v0 - m);
      p1[r] = __expf(v1 - m);
      float sum = p0[r] + p1[r];
#pragma unroll
      for (int off = 8; off > 0; off >>= 1) sum += __shfl_xor(sum, off, 16);
      l_run[r] = l_run[r] * alpha + sum;
      m_run[r] = m;
#pragma unroll
      for (int ct = 0; ct < 8; ++ct) O[ct][r] *= alpha;
    }

    // ---- transpose P (C layout -> A layout) via LDS ----
    __bf16* pw = &Ps[wave][0][0];
#pragma unroll
    for (int r = 0; r < 8; ++r) {
      const int m = r + 8 * hf;
      pw[m * 32 + nc]      = (__bf16)p0[r];
      pw[m * 32 + 16 + nc] = (__bf16)p1[r];
    }
    __syncthreads();  // P visible across lanes (also fences K/V reuse)

    // ---- O += P . V  (8 WMMA across the 128 head dims) ----
    const v16bf pf = frag_a(pw, 32, lane);
#pragma unroll
    for (int ct = 0; ct < 8; ++ct) {
      const v16bf vf = frag_b_pk(&VsPk[(ct * 16) * 8], 128 * 8, lane);
      O[ct] = WMMA_BF16(pf, vf, O[ct]);
    }
    __syncthreads();  // before next tile overwrites Ks/VsPk
  }

  // ---- normalize and write attention output [row, h*128 + d] as bf16 ----
  float invl[8];
#pragma unroll
  for (int r = 0; r < 8; ++r) invl[r] = 1.0f / l_run[r];
  __bf16* op = ao + (size_t)(b * SEQ + qbase) * DIM + h * DIM_HEAD;
#pragma unroll
  for (int ct = 0; ct < 8; ++ct)
#pragma unroll
    for (int r = 0; r < 8; ++r)
      op[(size_t)(r + 8 * hf) * DIM + ct * 16 + nc] = (__bf16)(O[ct][r] * invl[r]);
}

// ---------------------------------------------------------------------------
// Host-side launcher
// ---------------------------------------------------------------------------
extern "C" void kernel_launch(void* const* d_in, const int* in_sizes, int n_in,
                              void* d_out, int out_size, void* d_ws, size_t ws_size,
                              hipStream_t stream) {
  (void)in_sizes; (void)n_in; (void)out_size; (void)ws_size;
  const float* seq    = (const float*)d_in[0];
  const float* w_norm = (const float*)d_in[1];
  const float* w_qkv  = (const float*)d_in[2];
  const float* w_out  = (const float*)d_in[3];
  const float* pmem   = (const float*)d_in[4];
  float* out = (float*)d_out;

  // Workspace layout
  char* ws = (char*)d_ws;
  const size_t XN_BYTES  = (size_t)ROWS * DIM * sizeof(__bf16);        // 16 MB
  const size_t QKV_BYTES = (size_t)ROWS * QKV_COLS * sizeof(float);    // 100.7 MB
  const size_t HSEQ      = (size_t)BATCH * HEADS * SEQ * DIM_HEAD;     // elems per q/k/v

  __bf16* xn  = (__bf16*)ws;
  float*  qkv = (float*)(ws + XN_BYTES);
  __bf16* qr  = (__bf16*)(ws + XN_BYTES + QKV_BYTES);
  __bf16* kr  = qr + HSEQ;
  __bf16* vr  = kr + HSEQ;
  __bf16* ao  = (__bf16*)qkv;  // reuse qkv region after rope_repack consumed it

  // 1) RMSNorm -> bf16
  rmsnorm_kernel<<<ROWS, 256, 0, stream>>>(seq, w_norm, xn);
  // 2) QKV projection (WMMA bf16)
  gemm_bf16f32<QKV_COLS, DIM><<<dim3(QKV_COLS / 64, ROWS / 128), 256, 0, stream>>>(
      xn, w_qkv, qkv);
  // 3) Rotary + repack into per-head bf16 tensors
  rope_repack_kernel<<<dim3(SEQ, HEADS, BATCH), 64, 0, stream>>>(qkv, qr, kr, vr);
  // 4) Sliding-window attention with persistent tokens (WMMA bf16 + TDM)
  attn_kernel<<<dim3(SEQ / 128, HEADS, BATCH), 256, 0, stream>>>(qr, kr, vr, pmem, ao);
  // 5) Output projection (WMMA bf16) -> f32 result
  gemm_bf16f32<DIM, DIM><<<dim3(DIM / 64, ROWS / 128), 256, 0, stream>>>(
      ao, w_out, out);
}
